// spaTemBlock_4054449127576
// MI455X (gfx1250) — matrix operand order repaired and checked by reading, not compiled, not run on hardware
//
#include <hip/hip_runtime.h>

// ---------------------------------------------------------------------------
// Problem constants (from reference)
// ---------------------------------------------------------------------------
#define W_    12
#define N_    20000
#define C_    64
#define E_    320000
#define HOPS_ 2
#define KIN   (W_ * C_)          // 768 : im2col K dim for inception GEMM
#define CO2   128                // 64 sig + 64 gate output columns
#define G_ELEMS (W_ * CO2 * KIN) // 1,179,648 (f16 -> 2.36 MB) Toeplitz weights
#define WNC   (W_ * N_ * C_)     // 15,360,000 elements

typedef _Float16 v4h  __attribute__((ext_vector_type(4)));
typedef _Float16 v8h  __attribute__((ext_vector_type(8)));
typedef _Float16 v16h __attribute__((ext_vector_type(16)));
typedef float    v8f  __attribute__((ext_vector_type(8)));

// Build a v16h WMMA operand from two contiguous 16B chunks (lo -> e0..7, hi -> e8..15)
__device__ __forceinline__ v16h ld2x8h(const _Float16* lo, const _Float16* hi) {
  v8h a = *reinterpret_cast<const v8h*>(lo);
  v8h b = *reinterpret_cast<const v8h*>(hi);
  return __builtin_shufflevector(a, b, 0, 1, 2, 3, 4, 5, 6, 7,
                                       8, 9, 10, 11, 12, 13, 14, 15);
}

// D(16x16 f32) = A(16x32 f16) * B(32x16 f16) + C   (f32 accumulate)
__device__ __forceinline__ v8f wmma16(v16h a, v16h b, v8f c) {
  return __builtin_amdgcn_wmma_f32_16x16x32_f16(false, a, false, b, (short)0, c,
                                                false, false);
}

// ---------------------------------------------------------------------------
// 0) Convert x (W,N,C) f32 -> f16 shadow (one pass, 4 elems/thread)
// ---------------------------------------------------------------------------
__global__ void cvt_x_kernel(const float* __restrict__ x, _Float16* __restrict__ xh) {
  const size_t i = ((size_t)blockIdx.x * 256 + threadIdx.x) * 4;
  float4 v = *reinterpret_cast<const float4*>(x + i);
  v4h o = {(_Float16)v.x, (_Float16)v.y, (_Float16)v.z, (_Float16)v.w};
  *reinterpret_cast<v4h*>(xh + i) = o;
}

// ---------------------------------------------------------------------------
// 1) Build dense Toeplitz weight tensor G16[t][co2][kk] (f16, column-major in kk)
//    kk = s*64 + ci ; co2<64 -> signal channel, co2>=64 -> gate channel
// ---------------------------------------------------------------------------
__global__ void build_g_kernel(const float* __restrict__ sw1, const float* __restrict__ sw3,
                               const float* __restrict__ sw5, const float* __restrict__ sw7,
                               const float* __restrict__ gw1, const float* __restrict__ gw3,
                               const float* __restrict__ gw5, const float* __restrict__ gw7,
                               _Float16* __restrict__ G16) {
  int idx = blockIdx.x * 256 + threadIdx.x;
  if (idx >= G_ELEMS) return;
  int kk   = idx % KIN;
  int rest = idx / KIN;
  int co2  = rest & (CO2 - 1);
  int t    = rest >> 7;

  int s  = kk >> 6;
  int ci = kk & 63;
  int co = co2 & 63;
  int g  = co >> 4;         // kernel group: sizes 1,3,5,7
  int oc = co & 15;
  int ks = 2 * g + 1;
  int d  = s - t + g;       // tap index (padding p = g)

  float val = 0.0f;
  if (d >= 0 && d < ks) {
    const float* wp;
    if (co2 < 64) wp = (g == 0) ? sw1 : (g == 1) ? sw3 : (g == 2) ? sw5 : sw7;
    else          wp = (g == 0) ? gw1 : (g == 1) ? gw3 : (g == 2) ? gw5 : gw7;
    val = wp[(oc * C_ + ci) * ks + d];
  }
  G16[idx] = (_Float16)val;
}

// ---------------------------------------------------------------------------
// 2) Transpose+convert gcn_w (hop,w,k,col) f32 -> Wt (hop,w,col,k) f16
// ---------------------------------------------------------------------------
__global__ void build_wt_kernel(const float* __restrict__ gcn_w, _Float16* __restrict__ Wt) {
  int idx = blockIdx.x * 256 + threadIdx.x;   // < HOPS_*W_*64*64 = 98304
  int k   = idx & 63;
  int col = (idx >> 6) & 63;
  int ww  = idx >> 12;                        // hop*W_ + w
  Wt[((size_t)ww * C_ + col) * C_ + k] = (_Float16)gcn_w[((size_t)ww * C_ + k) * C_ + col];
}

// ---------------------------------------------------------------------------
// 3) Inception + gating: X(f16)[32 nodes x 768] @ G16[t][128 cols x 768]^T
//    Block: 256 threads / 8 waves; wave = col-tile (0..3 sig, 4..7 gate),
//    2 row-tiles per wave -> B fetched once, used by 2 WMMAs.
//    24 K-steps of v_wmma_f32_16x16x32_f16. Combine via LDS, write h (f16).
// ---------------------------------------------------------------------------
__global__ void inception_kernel(const _Float16* __restrict__ xh, const _Float16* __restrict__ G16,
                                 const float* __restrict__ sb1, const float* __restrict__ sb3,
                                 const float* __restrict__ sb5, const float* __restrict__ sb7,
                                 const float* __restrict__ gb1, const float* __restrict__ gb3,
                                 const float* __restrict__ gb5, const float* __restrict__ gb7,
                                 _Float16* __restrict__ hf16) {
  __shared__ float tl[16 * 256];   // 16 tiles of 16x16 f32

  const int t     = blockIdx.y;
  const int node0 = blockIdx.x * 32;
  const int wave  = threadIdx.x >> 5;
  const int lane  = threadIdx.x & 31;
  const int m16   = lane & 15;
  const int half  = lane >> 4;

  const int col2 = wave * 16 + m16;                        // B col in [0,128)
  const _Float16* Gcol = G16 + (size_t)(t * CO2 + col2) * KIN;

  const int n0 = node0 + m16;
  const int n1 = n0 + 16;

  v8f acc0 = {0.f, 0.f, 0.f, 0.f, 0.f, 0.f, 0.f, 0.f};
  v8f acc1 = {0.f, 0.f, 0.f, 0.f, 0.f, 0.f, 0.f, 0.f};

#pragma unroll 2
  for (int k0 = 0; k0 < KIN; k0 += 32) {
    const int s    = k0 >> 6;
    const int offA = (k0 & 63) + half * 8;   // both 8-chunks stay in this s-block
    const _Float16* xs = xh + (size_t)s * N_ * C_;
    const _Float16* a0p = xs + (size_t)n0 * C_ + offA;
    const _Float16* a1p = xs + (size_t)n1 * C_ + offA;
    v16h a0 = ld2x8h(a0p, a0p + 16);
    v16h a1 = ld2x8h(a1p, a1p + 16);
    const _Float16* bp = Gcol + k0 + half * 8;
    v16h b = ld2x8h(bp, bp + 16);
    acc0 = wmma16(a0, b, acc0);
    acc1 = wmma16(a1, b, acc1);
  }

  // C/D layout: element (v, lane) -> M = v + 8*half, N = lane&15
  {
    float* t0 = tl + (wave * 2 + 0) * 256;
    float* t1 = tl + (wave * 2 + 1) * 256;
    const int mbase = half << 3;
#pragma unroll
    for (int v = 0; v < 8; ++v) {
      t0[(mbase + v) * 16 + m16] = acc0[v];
      t1[(mbase + v) * 16 + m16] = acc1[v];
    }
  }
  __syncthreads();

  for (int i = threadIdx.x; i < 32 * C_; i += 256) {
    const int nr = i >> 6;            // node row in [0,32)
    const int co = i & 63;
    const int rt = nr >> 4;
    const int nl = nr & 15;
    const int g  = co >> 4;
    const int oc = co & 15;
    float sig  = tl[(((co >> 4)     ) * 2 + rt) * 256 + nl * 16 + (co & 15)];
    float gate = tl[(((co >> 4) + 4 ) * 2 + rt) * 256 + nl * 16 + (co & 15)];
    const float* sb = (g == 0) ? sb1 : (g == 1) ? sb3 : (g == 2) ? sb5 : sb7;
    const float* gb = (g == 0) ? gb1 : (g == 1) ? gb3 : (g == 2) ? gb5 : gb7;
    sig  += sb[oc];
    gate += gb[oc];
    float r  = fmaxf(sig, 0.0f);
    float sg = 1.0f / (1.0f + __expf(-gate));
    hf16[((size_t)t * N_ + node0 + nr) * C_ + co] = (_Float16)(r * sg);
  }
}

// ---------------------------------------------------------------------------
// 4) GCN dense step: z[w] = h16[w] @ gcn_w[hop,w]  (20000x64 @ 64x64, f16->f32)
//    Block: 8 waves = 2 row-tiles x 4 col-tiles; 2 K-steps per wave.
// ---------------------------------------------------------------------------
__global__ void gcn_gemm_kernel(const _Float16* __restrict__ hf16, const _Float16* __restrict__ Wt,
                                int hop, float* __restrict__ z) {
  const int w     = blockIdx.y;
  const int wave  = threadIdx.x >> 5;
  const int lane  = threadIdx.x & 31;
  const int m16   = lane & 15;
  const int half  = lane >> 4;
  const int node0 = blockIdx.x * 32 + (wave >> 2) * 16;
  const int col   = (wave & 3) * 16 + m16;

  const _Float16* A = hf16 + ((size_t)w * N_ + node0 + m16) * C_;
  const _Float16* B = Wt + (((size_t)hop * W_ + w) * C_ + col) * C_;

  v8f acc = {0.f, 0.f, 0.f, 0.f, 0.f, 0.f, 0.f, 0.f};
#pragma unroll
  for (int k0 = 0; k0 < C_; k0 += 32) {
    const _Float16* ap = A + k0 + half * 8;
    v16h a = ld2x8h(ap, ap + 16);
    const _Float16* bp = B + k0 + half * 8;
    v16h b = ld2x8h(bp, bp + 16);
    acc = wmma16(a, b, acc);
  }

  float* zrow = z + ((size_t)w * N_ + node0) * C_;
  const int mbase = half << 3;
#pragma unroll
  for (int v = 0; v < 8; ++v) zrow[(size_t)(mbase + v) * C_ + col] = acc[v];
}

// ---------------------------------------------------------------------------
// 5) zero / atomic scatter / elu+bias / final residual add
// ---------------------------------------------------------------------------
__global__ void zero_kernel(float4* __restrict__ p) {
  p[(size_t)blockIdx.x * 256 + threadIdx.x] = make_float4(0.f, 0.f, 0.f, 0.f);
}

// One wave per edge; each lane handles 2 channels (2 x global_atomic_add_f32).
__global__ void scatter_kernel(const float* __restrict__ z, const int* __restrict__ ei,
                               const float* __restrict__ ew, float* __restrict__ agg) {
  const int w    = blockIdx.y;
  const int wave = threadIdx.x >> 5;
  const int lane = threadIdx.x & 31;
  const int e    = blockIdx.x * 8 + wave;

  const int   src = ei[((size_t)w * 2 + 0) * E_ + e];
  const int   dst = ei[((size_t)w * 2 + 1) * E_ + e];
  const float wt  = ew[(size_t)w * E_ + e];

  const int c0 = lane * 2;
  const float2 v = *reinterpret_cast<const float2*>(z + ((size_t)w * N_ + src) * C_ + c0);
  float* ap = agg + ((size_t)w * N_ + dst) * C_ + c0;
  atomicAdd(ap + 0, wt * v.x);
  atomicAdd(ap + 1, wt * v.y);
}

// agg (f32) -> h = elu(agg + bias); keep f32 master and write f16 shadow.
__global__ void elu_bias_kernel(float* __restrict__ hf32, _Float16* __restrict__ hf16,
                                const float* __restrict__ gcn_b, int hop) {
  const int w = blockIdx.y;
  const int i = blockIdx.x * 256 + threadIdx.x;   // [0, N_*C_)
  const int c = i & 63;
  const float b = gcn_b[((size_t)hop * W_ + w) * C_ + c];
  const size_t off = (size_t)w * N_ * C_ + i;
  float v = hf32[off] + b;
  v = (v > 0.0f) ? v : (__expf(v) - 1.0f);
  hf32[off] = v;
  hf16[off] = (_Float16)v;
}

__global__ void add_x_kernel(const float4* __restrict__ hbuf, const float4* __restrict__ x,
                             float4* __restrict__ out) {
  const size_t i = (size_t)blockIdx.x * 256 + threadIdx.x;
  float4 a = hbuf[i];
  float4 b = x[i];
  out[i] = make_float4(a.x + b.x, a.y + b.y, a.z + b.z, a.w + b.w);
}

// ---------------------------------------------------------------------------
// Launch.  Workspace layout (bytes), with aliasing:
//   [0, 61.44MB)         : xh (30.72MB) + G16 (2.36MB)   -- dead after inception
//                          reused as z (f32, 61.44MB)    -- from first GEMM on
//   [61.44, 92.16MB)     : hf16 (30.72MB)
//   [92.16, 153.6MB)     : hf32 / agg (61.44MB)
//   [153.6MB, +196.6KB)  : Wt (f16 transposed gcn_w)
// Total ~153.8 MB.
// ---------------------------------------------------------------------------
extern "C" void kernel_launch(void* const* d_in, const int* in_sizes, int n_in,
                              void* d_out, int out_size, void* d_ws, size_t ws_size,
                              hipStream_t stream) {
  (void)in_sizes; (void)n_in; (void)out_size; (void)ws_size;

  const float* x   = (const float*)d_in[0];
  const int*   ei  = (const int*)d_in[1];
  const float* ew  = (const float*)d_in[2];
  const float* sw1 = (const float*)d_in[3];
  const float* sb1 = (const float*)d_in[4];
  const float* gw1 = (const float*)d_in[5];
  const float* gb1 = (const float*)d_in[6];
  const float* sw3 = (const float*)d_in[7];
  const float* sb3 = (const float*)d_in[8];
  const float* gw3 = (const float*)d_in[9];
  const float* gb3 = (const float*)d_in[10];
  const float* sw5 = (const float*)d_in[11];
  const float* sb5 = (const float*)d_in[12];
  const float* gw5 = (const float*)d_in[13];
  const float* gb5 = (const float*)d_in[14];
  const float* sw7 = (const float*)d_in[15];
  const float* sb7 = (const float*)d_in[16];
  const float* gw7 = (const float*)d_in[17];
  const float* gb7 = (const float*)d_in[18];
  const float* gcn_w = (const float*)d_in[19];
  const float* gcn_b = (const float*)d_in[20];

  char* ws = (char*)d_ws;
  const size_t XH_B   = (size_t)WNC * 2;        // 30,720,000
  const size_t Z_B    = (size_t)WNC * 4;        // 61,440,000
  const size_t HF16_B = (size_t)WNC * 2;

  _Float16* xh   = (_Float16*)(ws);             // aliased with z
  _Float16* G16  = (_Float16*)(ws + XH_B);      // aliased with z
  float*    z    = (float*)(ws);                // valid once inception is done
  _Float16* hf16 = (_Float16*)(ws + Z_B);
  float*    hf32 = (float*)(ws + Z_B + HF16_B); // agg buffer / f32 master h
  _Float16* Wt   = (_Float16*)(ws + Z_B + HF16_B + Z_B);

  cvt_x_kernel<<<WNC / (4 * 256), 256, 0, stream>>>(x, xh);
  build_g_kernel<<<G_ELEMS / 256, 256, 0, stream>>>(sw1, sw3, sw5, sw7,
                                                    gw1, gw3, gw5, gw7, G16);
  build_wt_kernel<<<(HOPS_ * W_ * C_ * C_) / 256, 256, 0, stream>>>(gcn_w, Wt);

  inception_kernel<<<dim3(N_ / 32, W_), 256, 0, stream>>>(
      xh, G16, sb1, sb3, sb5, sb7, gb1, gb3, gb5, gb7, hf16);

  for (int j = 0; j < HOPS_; ++j) {
    gcn_gemm_kernel<<<dim3(N_ / 32, W_), 256, 0, stream>>>(hf16, Wt, j, z);
    zero_kernel<<<WNC / (4 * 256), 256, 0, stream>>>((float4*)hf32);
    scatter_kernel<<<dim3(E_ / 8, W_), 256, 0, stream>>>(z, ei, ew, hf32);
    elu_bias_kernel<<<dim3(N_ * C_ / 256, W_), 256, 0, stream>>>(hf32, hf16, gcn_b, j);
  }

  add_x_kernel<<<WNC / (4 * 256), 256, 0, stream>>>(
      (const float4*)hf32, (const float4*)x, (float4*)d_out);
}
